// PaddedSoftmaxVSHCSA_13443247636577
// MI455X (gfx1250) — matrix operand order repaired and checked by reading, not compiled, not run on hardware
//
#include <hip/hip_runtime.h>
#include <hip/hip_bf16.h>

// ---------------------------------------------------------------------------
// Causal single-head attention (flash-attention style) for MI455X / gfx1250.
//   T=8192, C=128, NF=128.
//   - QKV projection: WMMA bf16 (24 v_wmma per wave) against pre-transposed W.
//   - Attention: flash loop, 16 v_wmma per 32-key tile per wave.
//   - K tiles via GLOBAL_LOAD_ASYNC_TO_LDS_B128 DMA, double-buffered so the
//     DMA for tile k+1 overlaps the WMMAs for tile k (ASYNCcnt drained only
//     at the iteration-end barrier).
// ---------------------------------------------------------------------------

#define TSEQ 8192
#define NC   128
#define NF   128
#define KB   32      // keys per block iteration
#define KT_PITCH 136 // bf16 elems per K-tile row  (272 B, 16B-aligned, bank-skewed)
#define VT_PITCH 48  // bf16 elems per Vt row      (96 B,  16B-aligned)
#define PB_PITCH 48  // bf16 elems per P row       (96 B,  16B-aligned)

#if defined(__has_builtin)
#  if __has_builtin(__builtin_amdgcn_global_load_async_to_lds_b128) && \
      __has_builtin(__builtin_amdgcn_s_wait_asynccnt)
#    define USE_ASYNC_LDS 1
#  endif
#endif
#ifndef USE_ASYNC_LDS
#  define USE_ASYNC_LDS 0
#endif

typedef __attribute__((ext_vector_type(16))) __bf16 v16bf;
typedef __attribute__((ext_vector_type(8)))  float  v8f;
typedef __attribute__((ext_vector_type(4)))  int    v4i;

union Frag {
  v16bf v;
  uint4 q[2];
  unsigned int u[8];
};

// hip_bf16 conversions: compile-safe everywhere, lowered to native packed
// cvt instructions when the target supports them (no manual RNE bit-twiddling).
__device__ __forceinline__ unsigned short f2bf(float f) {
  __hip_bfloat16 h = __float2bfloat16(f);
  return __builtin_bit_cast(unsigned short, h);
}
__device__ __forceinline__ unsigned int pack_bf2(float lo, float hi) {
  return (unsigned int)f2bf(lo) | ((unsigned int)f2bf(hi) << 16);
}

// ---------------------------------------------------------------------------
// Kernel 0: Wt[n][c] = bf16(W[c][n])  -- 384x128 bf16, B-fragment friendly.
// ---------------------------------------------------------------------------
__global__ void wconvert_kernel(const float* __restrict__ W,
                                unsigned short* __restrict__ Wt) {
  const int idx = blockIdx.x * blockDim.x + threadIdx.x;
  if (idx >= 3 * NF * NC) return;
  const int n = idx >> 7;          // output column 0..383
  const int c = idx & (NC - 1);    // contraction row 0..127
  Wt[n * NC + c] = f2bf(W[c * (3 * NF) + n]);
}

// ---------------------------------------------------------------------------
// Kernel 1: QKV projection on the matrix pipe.
//   512 blocks x 4 waves; each wave: 16 rows x 96 cols = 6 N-tiles x 4 K-chunks
//   = 24 v_wmma_f32_16x16x32_bf16.  x converted to bf16 A-frags on the fly.
// ---------------------------------------------------------------------------
__global__ void __launch_bounds__(128)
qkv_proj_wmma_kernel(const float* __restrict__ x,
                     const unsigned short* __restrict__ Wt,
                     const float* __restrict__ bias,
                     const int*   __restrict__ npadd,
                     unsigned short* __restrict__ qb,
                     unsigned short* __restrict__ kb,
                     unsigned short* __restrict__ vb) {
  const int tid  = threadIdx.x;
  const int wave = tid >> 5;
  const int lane = tid & 31;
  const int row  = lane & 15;
  const int hi   = lane >> 4;
  const int np   = npadd[0];
  const int rbase = blockIdx.x * 16;

  // A-fragments from x rows (f32 -> bf16). Layout: u[0..3] = K hi*8..+7,
  // u[4..7] = K 16+hi*8..+7 within each 32-wide chunk.
  Frag a[4];
#pragma unroll
  for (int ck = 0; ck < 4; ++ck) {
    const float* p = x + (rbase + row) * NC + ck * 32 + hi * 8;
#pragma unroll
    for (int e = 0; e < 4; ++e) {
      a[ck].u[e]     = pack_bf2(p[2 * e],      p[2 * e + 1]);
      a[ck].u[e + 4] = pack_bf2(p[16 + 2 * e], p[16 + 2 * e + 1]);
    }
  }

#pragma unroll
  for (int nt = 0; nt < 6; ++nt) {
    const int colbase = wave * 96 + nt * 16;
    v8f acc = (v8f)0.f;
#pragma unroll
    for (int ck = 0; ck < 4; ++ck) {
      // B layout: lane holds column (lane&15); K = hi*8..+7, 16+hi*8..+7.
      Frag bf;
      const unsigned short* p = Wt + (colbase + row) * NC + ck * 32 + hi * 8;
      bf.q[0] = *(const uint4*)p;
      bf.q[1] = *(const uint4*)(p + 16);
      acc = __builtin_amdgcn_wmma_f32_16x16x32_bf16(
                false, a[ck].v, false, bf.v, (short)0, acc, false, false);
    }
    const int   col  = colbase + row;      // C layout: N = lane&15
    const float bcol = bias[col];
    const int   which = col >> 7;          // 0:q 1:k 2:v
    const int   f     = col & (NF - 1);
#pragma unroll
    for (int r = 0; r < 8; ++r) {
      const int i = rbase + r + 8 * hi;    // C layout: M = r + 8*(lane>>4)
      const unsigned short e = f2bf(acc[r] + bcol);
      if (which == 0)      qb[i * NF + f] = e;
      else if (which == 1) kb[i * NF + f] = e;
      else                 vb[i * NF + f] = (i >= np) ? e : (unsigned short)0;
    }
  }
}

// ---------------------------------------------------------------------------
// Tile staging helpers (flash kernel).
// ---------------------------------------------------------------------------
__device__ __forceinline__ void stage_k_tile(const unsigned short* __restrict__ K,
                                             int kblk, unsigned short* dst, int tid) {
#if USE_ASYNC_LDS
  for (int it = tid; it < KB * 16; it += 128) {          // 8-elem chunks
    const int r  = it >> 4;
    const int cc = (it & 15) << 3;
    unsigned short* gp =
        const_cast<unsigned short*>(&K[(kblk * KB + r) * NF + cc]);
    __builtin_amdgcn_global_load_async_to_lds_b128(
        (__attribute__((address_space(1))) v4i*)gp,
        (__attribute__((address_space(3))) v4i*)&dst[r * KT_PITCH + cc],
        0, 0);
  }
#else
  for (int it = tid; it < KB * 16; it += 128) {
    const int r  = it >> 4;
    const int cc = (it & 15) << 3;
    *(uint4*)&dst[r * KT_PITCH + cc] =
        *(const uint4*)&K[(kblk * KB + r) * NF + cc];
  }
#endif
}

__device__ __forceinline__ void stage_v_tile(const unsigned short* __restrict__ V,
                                             int kblk, unsigned short* dst, int tid) {
  // transposed: dst[n][k] = V[kblk*32+k][n]
  for (int it = tid; it < KB * (NF / 2); it += 128) {    // 2-elem chunks
    const int k  = it >> 6;
    const int n2 = (it & 63) << 1;
    const unsigned int d = *(const unsigned int*)&V[(kblk * KB + k) * NF + n2];
    dst[n2 * VT_PITCH + k]       = (unsigned short)(d & 0xFFFFu);
    dst[(n2 + 1) * VT_PITCH + k] = (unsigned short)(d >> 16);
  }
}

// ---------------------------------------------------------------------------
// Kernel 2: flash attention.  4 waves/block, 16 query rows per wave.
// Double-buffered 32-key tiles: DMA for tile k+1 overlaps WMMAs for tile k.
// ---------------------------------------------------------------------------
__global__ void __launch_bounds__(128)
flash_attn_kernel(const unsigned short* __restrict__ Q,
                  const unsigned short* __restrict__ K,
                  const unsigned short* __restrict__ V,
                  const int* __restrict__ npadd,
                  float* __restrict__ out) {
  __shared__ unsigned short kt[2][KB * KT_PITCH];   // K tiles, row-major [32][136]
  __shared__ unsigned short vt[2][NF * VT_PITCH];   // V tiles, TRANSPOSED [128][48]
  __shared__ unsigned short pb[4][16 * PB_PITCH];   // per-wave P scratch [16][48]

  const int tid  = threadIdx.x;
  const int wave = tid >> 5;
  const int lane = tid & 31;
  const int row  = lane & 15;
  const int hi   = lane >> 4;
  const int np   = npadd[0];

  const int qbase = blockIdx.x * 64 + wave * 16;
  const int qhi   = qbase + 15;

  // ---- Q A-fragments (register resident across the whole key loop) --------
  Frag a[4];
#pragma unroll
  for (int ck = 0; ck < 4; ++ck) {
    const unsigned short* p = Q + (qbase + row) * NF + ck * 32 + hi * 8;
    a[ck].q[0] = *(const uint4*)(p);
    a[ck].q[1] = *(const uint4*)(p + 16);
  }

  v8f o[8];
#pragma unroll
  for (int t = 0; t < 8; ++t) o[t] = (v8f)0.f;
  float mrow[8], lrow[8];
#pragma unroll
  for (int r = 0; r < 8; ++r) { mrow[r] = -__builtin_inff(); lrow[r] = 0.f; }

  const int kbmax = (blockIdx.x * 64 + 63) >> 5;   // inclusive, block-wide

  // ---- prologue: stage tile 0 into buffer 0 -------------------------------
  stage_k_tile(K, 0, kt[0], tid);
  stage_v_tile(V, 0, vt[0], tid);
#if USE_ASYNC_LDS
  __builtin_amdgcn_s_wait_asynccnt(0);
#endif
  __syncthreads();

  for (int kblk = 0; kblk <= kbmax; ++kblk) {
    const int p = kblk & 1;

    // ---- kick off DMA / staging for tile k+1 into the other buffer --------
    // (buffer p^1 was last read in iteration k-1; the end-of-iteration
    //  barrier below protects it.)
    if (kblk < kbmax) {
      stage_k_tile(K, kblk + 1, kt[p ^ 1], tid);
      stage_v_tile(V, kblk + 1, vt[p ^ 1], tid);
    }
    if (kblk + 2 <= kbmax && lane == 0) {
      __builtin_prefetch(&K[(kblk + 2) * KB * NF], 0, 1);
      __builtin_prefetch(&V[(kblk + 2) * KB * NF], 0, 1);
    }

    // causal / padding skip at tile granularity (all waves still hit barriers)
    const bool active = (kblk * KB <= qhi) && (kblk * KB + KB > np);
    if (active) {
      const unsigned short* ktp = kt[p];
      const unsigned short* vtp = vt[p];

      // ---- S = (Q Kt) * scale : two 16x16 column tiles --------------------
      v8f sAcc[2];
#pragma unroll
      for (int nt = 0; nt < 2; ++nt) {
        v8f s = (v8f)0.f;
#pragma unroll
        for (int ck = 0; ck < 4; ++ck) {
          Frag bf;
          const unsigned short* pk =
              &ktp[(nt * 16 + row) * KT_PITCH + ck * 32 + hi * 8];
          bf.q[0] = *(const uint4*)pk;
          bf.q[1] = *(const uint4*)(pk + 16);
          s = __builtin_amdgcn_wmma_f32_16x16x32_bf16(
                  false, a[ck].v, false, bf.v, (short)0, s, false, false);
        }
        sAcc[nt] = s;
      }

      // ---- online softmax over the full 16x32 score tile ------------------
      const int j0 = kblk * KB + row;
      const int j1 = j0 + 16;
#pragma unroll
      for (int r = 0; r < 8; ++r) {
        const int i = qbase + r + 8 * hi;
        float s0 = sAcc[0][r] * 0.08838834764831845f;   // 1/sqrt(128)
        float s1 = sAcc[1][r] * 0.08838834764831845f;
        if (j0 > i || j0 < np || i < np) s0 = -__builtin_inff();
        if (j1 > i || j1 < np || i < np) s1 = -__builtin_inff();

        float rm = fmaxf(s0, s1);
        rm = fmaxf(rm, __shfl_xor(rm, 1, 32));
        rm = fmaxf(rm, __shfl_xor(rm, 2, 32));
        rm = fmaxf(rm, __shfl_xor(rm, 4, 32));
        rm = fmaxf(rm, __shfl_xor(rm, 8, 32));
        const float newm = fmaxf(mrow[r], rm);
        const float base = (newm == -__builtin_inff()) ? 0.f : newm;
        const float corr = __expf(mrow[r] - base);      // 0 if mrow == -inf
        const float p0 = __expf(s0 - base);             // 0 if masked
        const float p1 = __expf(s1 - base);

        float rs = p0 + p1;
        rs += __shfl_xor(rs, 1, 32);
        rs += __shfl_xor(rs, 2, 32);
        rs += __shfl_xor(rs, 4, 32);
        rs += __shfl_xor(rs, 8, 32);
        lrow[r] = lrow[r] * corr + rs;
        mrow[r] = newm;
#pragma unroll
        for (int t = 0; t < 8; ++t) o[t][r] = o[t][r] * corr;

        pb[wave][(r + 8 * hi) * PB_PITCH + row]      = f2bf(p0);
        pb[wave][(r + 8 * hi) * PB_PITCH + 16 + row] = f2bf(p1);
      }

      // ---- O += P (16x32) @ V (32x128) : 8 WMMAs --------------------------
      // DS ops from the same wave are in-order, so the reload sees the stores.
      Frag pa;
      {
        const unsigned short* pp = &pb[wave][row * PB_PITCH + hi * 8];
        pa.q[0] = *(const uint4*)pp;
        pa.q[1] = *(const uint4*)(pp + 16);
      }
#pragma unroll
      for (int ot = 0; ot < 8; ++ot) {
        Frag vf;
        const unsigned short* vp = &vtp[(ot * 16 + row) * VT_PITCH + hi * 8];
        vf.q[0] = *(const uint4*)vp;
        vf.q[1] = *(const uint4*)(vp + 16);
        o[ot] = __builtin_amdgcn_wmma_f32_16x16x32_bf16(
                    false, pa.v, false, vf.v, (short)0, o[ot], false, false);
      }
    }

    // ---- drain this wave's DMA for tile k+1, then publish buffers ---------
#if USE_ASYNC_LDS
    __builtin_amdgcn_s_wait_asynccnt(0);
#endif
    __syncthreads();
  }

  // ---- epilogue: normalize and store --------------------------------------
#pragma unroll
  for (int r = 0; r < 8; ++r) {
    const int i = qbase + r + 8 * hi;
    const float inv = (lrow[r] > 0.f) ? (1.f / lrow[r]) : 0.f;
#pragma unroll
    for (int ot = 0; ot < 8; ++ot)
      out[i * NF + ot * 16 + row] = o[ot][r] * inv;
  }
}

// ---------------------------------------------------------------------------
extern "C" void kernel_launch(void* const* d_in, const int* in_sizes, int n_in,
                              void* d_out, int out_size, void* d_ws, size_t ws_size,
                              hipStream_t stream) {
  const float* x  = (const float*)d_in[0];
  const float* W  = (const float*)d_in[1];
  const float* b  = (const float*)d_in[2];
  const int*   np = (const int*)d_in[3];
  float* out = (float*)d_out;

  unsigned short* qb = (unsigned short*)d_ws;            // [T][NF] bf16
  unsigned short* kb = qb + (size_t)TSEQ * NF;           // [T][NF] bf16
  unsigned short* vb = kb + (size_t)TSEQ * NF;           // [T][NF] bf16
  unsigned short* wt = vb + (size_t)TSEQ * NF;           // [384][128] bf16

  wconvert_kernel<<<(3 * NF * NC + 255) / 256, 256, 0, stream>>>(W, wt);
  qkv_proj_wmma_kernel<<<TSEQ / 16, 128, 0, stream>>>(x, wt, b, np, qb, kb, vb);
  flash_attn_kernel<<<TSEQ / 64, 128, 0, stream>>>(qb, kb, vb, np, out);
}
